// BipartiteGraphAttentionLayer_87668872446040
// MI455X (gfx1250) — matrix operand order repaired and checked by reading, not compiled, not run on hardware
//
#include <hip/hip_runtime.h>
#include <math.h>

typedef __attribute__((ext_vector_type(16))) __bf16         v16bf;
typedef __attribute__((ext_vector_type(8)))  float          v8f;
typedef __attribute__((ext_vector_type(16))) unsigned short v16us;

#define IN_DIM   256
#define OUT_DIM  128
#define LEAKY    0.2f

// Split an f32 into bf16 hi (truncated) and bf16 lo (residual). hi+lo captures
// ~16 mantissa bits; bf16x3 WMMA then gives ~1e-5 relative GEMM error.
__device__ __forceinline__ void f32_to_bf16x2(float f, unsigned short& hi, unsigned short& lo) {
    unsigned u = __float_as_uint(f);
    hi = (unsigned short)(u >> 16);
    float fhi = __uint_as_float(u & 0xFFFF0000u);
    float flo = f - fhi;                      // exact (<=16 significant bits)
    lo = (unsigned short)(__float_as_uint(flo) >> 16);
}

// ---------------------------------------------------------------------------
// Zero the output accumulator + softmax denominator each launch (graph-safe).
__global__ void zero_kernel(float* __restrict__ a, size_t na,
                            float* __restrict__ b, size_t nb) {
    size_t i = (size_t)blockIdx.x * blockDim.x + threadIdx.x;
    size_t stride = (size_t)gridDim.x * blockDim.x;
    for (size_t j = i; j < na; j += stride) a[j] = 0.0f;
    for (size_t j = i; j < nb; j += stride) b[j] = 0.0f;
}

// ---------------------------------------------------------------------------
// W [K=256][N=128] f32  ->  WT_hi/WT_lo [N=128][K=256] bf16 (bits in ushort).
__global__ void convert_w_kernel(const float* __restrict__ W,
                                 unsigned short* __restrict__ wt_hi,
                                 unsigned short* __restrict__ wt_lo) {
    int idx = blockIdx.x * blockDim.x + threadIdx.x;
    if (idx >= IN_DIM * OUT_DIM) return;
    int n = idx >> 8;          // 0..127 output column (row of WT)
    int k = idx & 255;         // 0..255
    float f = W[k * OUT_DIM + n];
    unsigned short h, l;
    f32_to_bf16x2(f, h, l);
    wt_hi[n * IN_DIM + k] = h;
    wt_lo[n * IN_DIM + k] = l;
}

// ---------------------------------------------------------------------------
// H[M][128] = X[M][256] @ W[256][128] via bf16x3 split WMMA.
// One wave per 16-row M tile; 8 N-tiles of 16 cols; K loop 8 x 32.
__global__ __launch_bounds__(256) void gemm_bf16x3_kernel(
    const float* __restrict__ X,
    const unsigned short* __restrict__ wt_hi_us,
    const unsigned short* __restrict__ wt_lo_us,
    float* __restrict__ H, int M) {
    const __bf16* __restrict__ WT_hi = (const __bf16*)wt_hi_us;
    const __bf16* __restrict__ WT_lo = (const __bf16*)wt_lo_us;

    const int wave = threadIdx.x >> 5;
    const int lane = threadIdx.x & 31;
    const int m0 = blockIdx.x * 128 + wave * 16;
    if (m0 >= M) return;                 // wave-uniform: EXEC stays all-ones

    const int row  = lane & 15;          // A row within tile / B column within tile
    const int half = lane >> 4;          // K-half selector per ISA 16-bit layouts
    int m = m0 + row;
    if (m > M - 1) m = M - 1;            // clamp tail reads (stores guarded below)
    const float* __restrict__ xrow = X + (size_t)m * IN_DIM;

    v8f acc[8] = {};

#pragma unroll
    for (int k0 = 0; k0 < IN_DIM; k0 += 32) {
        // A fragment (this code's contiguous k order, not ISA table order): lane holds row m, 16 consecutive K
        // starting at k0+16*half (matches B).
        const float4 q0 = *(const float4*)(xrow + k0 + 16 * half);
        const float4 q1 = *(const float4*)(xrow + k0 + 16 * half + 4);
        const float4 q2 = *(const float4*)(xrow + k0 + 16 * half + 8);
        const float4 q3 = *(const float4*)(xrow + k0 + 16 * half + 12);
        float av[16];
        av[0] = q0.x;  av[1] = q0.y;  av[2]  = q0.z;  av[3]  = q0.w;
        av[4] = q1.x;  av[5] = q1.y;  av[6]  = q1.z;  av[7]  = q1.w;
        av[8] = q2.x;  av[9] = q2.y;  av[10] = q2.z;  av[11] = q2.w;
        av[12] = q3.x; av[13] = q3.y; av[14] = q3.z;  av[15] = q3.w;

        v16us ah, al;
#pragma unroll
        for (int i = 0; i < 16; ++i) {
            unsigned short h, l;
            f32_to_bf16x2(av[i], h, l);
            ah[i] = h;
            al[i] = l;
        }
        v16bf Ahi = __builtin_bit_cast(v16bf, ah);
        v16bf Alo = __builtin_bit_cast(v16bf, al);

#pragma unroll
        for (int t = 0; t < 8; ++t) {
            // B fragment (contiguous k order, same as A): lane c = col 16t+row, 16 consecutive
            // K values starting at k0 + 16*half  -> contiguous 32B per lane.
            size_t boff = (size_t)(t * 16 + row) * IN_DIM + k0 + 16 * half;
            v16bf Bhi = *(const v16bf*)(WT_hi + boff);
            v16bf Blo = *(const v16bf*)(WT_lo + boff);
            acc[t] = __builtin_amdgcn_wmma_f32_16x16x32_bf16(
                false, Ahi, false, Bhi, (short)0, acc[t], false, false);
            acc[t] = __builtin_amdgcn_wmma_f32_16x16x32_bf16(
                false, Ahi, false, Blo, (short)0, acc[t], false, false);
            acc[t] = __builtin_amdgcn_wmma_f32_16x16x32_bf16(
                false, Alo, false, Bhi, (short)0, acc[t], false, false);
        }
    }

    // C/D layout: lane l, VGPR r -> row = 8*(l>>4)+r, col = l&15 (per n-tile).
    const int rbase = half * 8;
#pragma unroll
    for (int t = 0; t < 8; ++t) {
#pragma unroll
        for (int r = 0; r < 8; ++r) {
            int mm = m0 + rbase + r;
            if (mm < M) H[(size_t)mm * OUT_DIM + t * 16 + row] = acc[t][r];
        }
    }
}

// ---------------------------------------------------------------------------
// score[row] = dot(H[row, :], avec[0:128]); one wave per row (wave32 reduce).
__global__ void score_kernel(const float* __restrict__ H,
                             const float* __restrict__ avec,
                             float* __restrict__ score, int M) {
    int wave = threadIdx.x >> 5, lane = threadIdx.x & 31;
    int r = blockIdx.x * 8 + wave;
    if (r >= M) return;
    const float4 h4 = *(const float4*)(H + (size_t)r * OUT_DIM + lane * 4);
    const float4 a4 = *(const float4*)(avec + lane * 4);
    float s = h4.x * a4.x + h4.y * a4.y + h4.z * a4.z + h4.w * a4.w;
#pragma unroll
    for (int m = 1; m < 32; m <<= 1) s += __shfl_xor(s, m, 32);
    if (lane == 0) score[r] = s;
}

// ---------------------------------------------------------------------------
// Pass 1 over edges: e = clip(leaky_relu(si+sj)), exp, segment-sum into denom.
__global__ void edge_exp_kernel(const int* __restrict__ src_id,
                                const int* __restrict__ tgt_id,
                                const float* __restrict__ score_i,
                                const float* __restrict__ score_j,
                                float* __restrict__ exp_e,
                                float* __restrict__ denom, int E) {
    int e = blockIdx.x * blockDim.x + threadIdx.x;
    if (e >= E) return;
    int s = src_id[e], t = tgt_id[e];
    float x = score_i[s] + score_j[t];
    float lr = x > 0.0f ? x : LEAKY * x;
    lr = fminf(fmaxf(lr, -30.0f), 30.0f);
    float ex = __expf(lr);
    exp_e[e] = ex;
    unsafeAtomicAdd(&denom[s], ex);
}

// ---------------------------------------------------------------------------
// Pass 2: one wave per edge; lane moves 4 floats of h_j[t]*alpha into out[s].
// h_j (51MB) and out (51MB) are L2-resident on MI455X (192MB L2).
__global__ void edge_msg_kernel(const int* __restrict__ src_id,
                                const int* __restrict__ tgt_id,
                                const float* __restrict__ h_j,
                                const float* __restrict__ exp_e,
                                const float* __restrict__ denom,
                                float* __restrict__ out, int E) {
    int wave = threadIdx.x >> 5, lane = threadIdx.x & 31;
    int e = blockIdx.x * 8 + wave;
    if (e >= E) return;
    int s = src_id[e], t = tgt_id[e];
    float alpha = exp_e[e] / (denom[s] + 1e-8f);
    const float4 v = *(const float4*)(h_j + (size_t)t * OUT_DIM + lane * 4);
    float* o = out + (size_t)s * OUT_DIM + lane * 4;
    unsafeAtomicAdd(o + 0, v.x * alpha);
    unsafeAtomicAdd(o + 1, v.y * alpha);
    unsafeAtomicAdd(o + 2, v.z * alpha);
    unsafeAtomicAdd(o + 3, v.w * alpha);
}

// ---------------------------------------------------------------------------
__global__ void elu_kernel(float* __restrict__ out, size_t n) {
    size_t i = (size_t)blockIdx.x * blockDim.x + threadIdx.x;
    size_t stride = (size_t)gridDim.x * blockDim.x;
    for (size_t j = i; j < n; j += stride) {
        float x = out[j];
        out[j] = x > 0.0f ? x : expm1f(x);
    }
}

// ---------------------------------------------------------------------------
extern "C" void kernel_launch(void* const* d_in, const int* in_sizes, int n_in,
                              void* d_out, int out_size, void* d_ws, size_t ws_size,
                              hipStream_t stream) {
    const float* src   = (const float*)d_in[0];
    const float* tgt   = (const float*)d_in[1];
    const float* W_src = (const float*)d_in[2];
    const float* W_tgt = (const float*)d_in[3];
    const float* avec  = (const float*)d_in[4];   // 256 floats
    const int*   eidx  = (const int*)d_in[5];     // [2][E]

    const int n_src = in_sizes[0] / IN_DIM;
    const int n_tgt = in_sizes[1] / IN_DIM;
    const int E     = in_sizes[5] / 2;
    const int* src_id = eidx;
    const int* tgt_id = eidx + E;

    // Workspace carve (all offsets kept 64B-aligned).
    float* ws = (float*)d_ws;
    size_t off = 0;
    auto alloc_f = [&](size_t n) { float* p = ws + off; off += (n + 15) & ~(size_t)15; return p; };
    float* h_i     = alloc_f((size_t)n_src * OUT_DIM);
    float* h_j     = alloc_f((size_t)n_tgt * OUT_DIM);
    float* score_i = alloc_f(n_src);
    float* score_j = alloc_f(n_tgt);
    float* denom   = alloc_f(n_src);
    float* exp_e   = alloc_f(E);
    unsigned short* wt = (unsigned short*)(ws + off);  // 4 x [128][256] bf16-bit arrays
    unsigned short* wt_src_hi = wt;
    unsigned short* wt_src_lo = wt + 1 * IN_DIM * OUT_DIM;
    unsigned short* wt_tgt_hi = wt + 2 * IN_DIM * OUT_DIM;
    unsigned short* wt_tgt_lo = wt + 3 * IN_DIM * OUT_DIM;

    float* out = (float*)d_out;

    // 1) zero accumulator + denom (ws/out are poisoned, not re-zeroed by harness)
    zero_kernel<<<2048, 256, 0, stream>>>(out, (size_t)out_size, denom, (size_t)n_src);

    // 2) weight convert/transpose to bf16 hi/lo
    convert_w_kernel<<<(IN_DIM * OUT_DIM + 255) / 256, 256, 0, stream>>>(W_src, wt_src_hi, wt_src_lo);
    convert_w_kernel<<<(IN_DIM * OUT_DIM + 255) / 256, 256, 0, stream>>>(W_tgt, wt_tgt_hi, wt_tgt_lo);

    // 3) GEMMs (WMMA bf16x3)
    gemm_bf16x3_kernel<<<(n_src + 127) / 128, 256, 0, stream>>>(src, wt_src_hi, wt_src_lo, h_i, n_src);
    gemm_bf16x3_kernel<<<(n_tgt + 127) / 128, 256, 0, stream>>>(tgt, wt_tgt_hi, wt_tgt_lo, h_j, n_tgt);

    // 4) attention scores
    score_kernel<<<(n_src + 7) / 8, 256, 0, stream>>>(h_i, avec, score_i, n_src);
    score_kernel<<<(n_tgt + 7) / 8, 256, 0, stream>>>(h_j, avec + OUT_DIM, score_j, n_tgt);

    // 5) edge softmax numerators + denominators
    edge_exp_kernel<<<(E + 255) / 256, 256, 0, stream>>>(src_id, tgt_id, score_i, score_j,
                                                         exp_e, denom, E);

    // 6) weighted message scatter
    edge_msg_kernel<<<(E + 7) / 8, 256, 0, stream>>>(src_id, tgt_id, h_j, exp_e, denom, out, E);

    // 7) ELU in place
    elu_kernel<<<2048, 256, 0, stream>>>(out, (size_t)out_size);
}
